// LSTMTree_14972255994561
// MI455X (gfx1250) — matrix operand order repaired and checked by reading, compiled, not run on hardware
//
#include <hip/hip_runtime.h>

#define H 128
#define NL 65536
#define NLEVELS 16

typedef __attribute__((ext_vector_type(16))) _Float16 v16h;
typedef __attribute__((ext_vector_type(8)))  _Float16 v8h;
typedef __attribute__((ext_vector_type(8)))  float    v8f;

union AFrag { v16h v; v8h h[2]; };

#define WMMA(acc, a, b) \
  acc = __builtin_amdgcn_wmma_f32_16x16x32_f16(false, (a), false, (b), (short)0, (acc), false, false)

// ---- fragment loaders -------------------------------------------------------

// A-fragment (16x32 f16, MxK): lane m (0..15) holds K = kc0..kc0+7 and kc0+16..kc0+23
// where kc0 = kk + (lane>=16 ? 8 : 0). Row pointer is per-lane (row = lane%16).
__device__ __forceinline__ v16h load_a16(const _Float16* __restrict__ row, int kc0) {
  AFrag a;
  a.h[0] = *(const v8h*)(row + kc0);
  a.h[1] = *(const v8h*)(row + kc0 + 16);
  return a.v;
}

// Same but over the virtual concat xc = [hl | hr] (K = 256). Each 8-half chunk
// lies entirely inside one child's row (chunks are 8-aligned, 128 % 32 == 0).
__device__ __forceinline__ v16h load_a_cat(const _Float16* __restrict__ hl,
                                           const _Float16* __restrict__ hr, int kc0) {
  AFrag a;
  const int kc1 = kc0 + 16;
  const _Float16* p0 = (kc0 < H) ? (hl + kc0) : (hr + (kc0 - H));
  const _Float16* p1 = (kc1 < H) ? (hl + kc1) : (hr + (kc1 - H));
  a.h[0] = *(const v8h*)p0;
  a.h[1] = *(const v8h*)p1;
  return a.v;
}

// A-fragment from an f32 row (embedding gather), converted on the fly.
__device__ __forceinline__ v16h load_a_f32(const float* __restrict__ row, int kc0) {
  AFrag a;
  const float* p0 = row + kc0;
  const float* p1 = row + kc0 + 16;
#pragma unroll
  for (int j = 0; j < 8; ++j) {
    a.h[0][j] = (_Float16)p0[j];
    a.h[1][j] = (_Float16)p1[j];
  }
  return a.v;
}

// B-fragment (Kx16 f16): lane n (0..15) holds W[n, kb..kb+7] and W[n, kb+16..kb+23],
// kb = kk + (lane>=16 ? 8 : 0). wp already points at W + n*ld + kb.
__device__ __forceinline__ v16h load_b16(const _Float16* __restrict__ wp) {
  AFrag b;
  b.h[0] = *(const v8h*)(wp);
  b.h[1] = *(const v8h*)(wp + 16);
  return b.v;
}

// ---- math -------------------------------------------------------------------

__device__ __forceinline__ float fsigmoid(float x) {
  return 1.0f / (1.0f + __expf(-x));
}
__device__ __forceinline__ float ftanh(float x) {
  x = fminf(fmaxf(x, -15.0f), 15.0f);
  float e = __expf(2.0f * x);
  return (e - 1.0f) / (e + 1.0f);
}

// ---- small kernels ----------------------------------------------------------

__global__ void cvt_f32_f16_kernel(const float* __restrict__ src,
                                   _Float16* __restrict__ dst, int n) {
  int i = blockIdx.x * 256 + threadIdx.x;
  if (i < n) dst[i] = (_Float16)src[i];
}

__global__ void copy_labels_kernel(const int* __restrict__ labels,
                                   float* __restrict__ dst, int n) {
  int i = blockIdx.x * 256 + threadIdx.x;
  if (i < n) dst[i] = (float)labels[i];
}

// ---- leaf stage -------------------------------------------------------------
// x = embed[tok]; i,o = sigmoid(x@W^T+b); u = tanh(...); c = i*u;
// h = o*tanh(c); out = h@out_w^T + out_b.
__global__ __launch_bounds__(256) void tree_leaf_kernel(
    const int* __restrict__ tokens, const float* __restrict__ embed,
    _Float16* __restrict__ h_out, float* __restrict__ c_out,
    float* __restrict__ out,
    const _Float16* __restrict__ Wi, const _Float16* __restrict__ Wo,
    const _Float16* __restrict__ Wu,
    const float* __restrict__ Wi_b, const float* __restrict__ Wo_b,
    const float* __restrict__ Wu_b,
    const float* __restrict__ ow, const float* __restrict__ ob, int n) {
  __shared__ float h_lds[16 * H];
  const int tid  = threadIdx.x;
  const int wave = tid >> 5;
  const int lane = tid & 31;
  const int lrow = lane & 15;
  const int hi   = lane >> 4;
  const int mbase = blockIdx.x * 16;
  const int col = (wave << 4) + lrow;

  int mrow = mbase + lrow;
  if (mrow >= n) mrow = n - 1;
  const float* xrow = embed + (size_t)tokens[mrow] * H;

  v8f acc_i, acc_o, acc_u;
  {
    float bi = Wi_b[col], bo = Wo_b[col], bu = Wu_b[col];
#pragma unroll
    for (int r = 0; r < 8; ++r) { acc_i[r] = bi; acc_o[r] = bo; acc_u[r] = bu; }
  }

#pragma unroll
  for (int kk = 0; kk < H; kk += 32) {
    const int kc0 = kk + hi * 8;
    const int kb  = kk + hi * 8;
    v16h a  = load_a_f32(xrow, kc0);
    v16h bi = load_b16(Wi + (size_t)col * H + kb);
    v16h bo = load_b16(Wo + (size_t)col * H + kb);
    v16h bu = load_b16(Wu + (size_t)col * H + kb);
    WMMA(acc_i, a, bi);
    WMMA(acc_o, a, bo);
    WMMA(acc_u, a, bu);
  }

#pragma unroll
  for (int r = 0; r < 8; ++r) {
    const int lm = r + 8 * hi;           // local row 0..15
    const int mm = mbase + lm;           // node index
    float iv = fsigmoid(acc_i[r]);
    float ov = fsigmoid(acc_o[r]);
    float uv = ftanh(acc_u[r]);
    float cv = iv * uv;
    float hv = ov * ftanh(cv);
    if (mm < n) {
      c_out[(size_t)mm * H + col] = cv;
      h_out[(size_t)mm * H + col] = (_Float16)hv;
    }
    h_lds[lm * H + col] = hv;
  }
  __syncthreads();

  if (tid < 80) {
    const int node = tid / 5, cls = tid % 5;
    const int mm = mbase + node;
    if (mm < n) {
      float s = ob[cls];
      const float* wr = ow + cls * H;
#pragma unroll 8
      for (int k = 0; k < H; ++k) s += h_lds[node * H + k] * wr[k];
      out[(size_t)mm * 5 + cls] = s;
    }
  }
}

// ---- one big merge level (n_out >= 32): children 2m, 2m+1 -> node m --------
__global__ __launch_bounds__(256) void tree_level_kernel(
    const _Float16* __restrict__ h_in, const float* __restrict__ c_in,
    _Float16* __restrict__ h_out, float* __restrict__ c_out,
    float* __restrict__ out,
    const _Float16* __restrict__ Ui, const _Float16* __restrict__ Uo,
    const _Float16* __restrict__ Uu,
    const _Float16* __restrict__ F1, const _Float16* __restrict__ F2,
    const float* __restrict__ Ui_b, const float* __restrict__ Uo_b,
    const float* __restrict__ Uu_b,
    const float* __restrict__ F1_b, const float* __restrict__ F2_b,
    const float* __restrict__ ow, const float* __restrict__ ob, int n_out) {
  __shared__ float h_lds[16 * H];
  const int tid  = threadIdx.x;
  const int wave = tid >> 5;
  const int lane = tid & 31;
  const int lrow = lane & 15;
  const int hi   = lane >> 4;
  const int mbase = blockIdx.x * 16;
  const int col = (wave << 4) + lrow;

  int mrow = mbase + lrow;
  if (mrow >= n_out) mrow = n_out - 1;
  const _Float16* hl_row = h_in + (size_t)(2 * mrow) * H;
  const _Float16* hr_row = hl_row + H;

  v8f acc_i, acc_o, acc_u, acc_f1, acc_f2;
  {
    float bi = Ui_b[col], bo = Uo_b[col], bu = Uu_b[col];
    float b1 = F1_b[col], b2 = F2_b[col];
#pragma unroll
    for (int r = 0; r < 8; ++r) {
      acc_i[r] = bi; acc_o[r] = bo; acc_u[r] = bu;
      acc_f1[r] = b1; acc_f2[r] = b2;
    }
  }

  // U gates over xc = [hl | hr], K = 256; one A-fragment reused for 3 WMMAs.
#pragma unroll
  for (int kk = 0; kk < 2 * H; kk += 32) {
    const int kc0 = kk + hi * 8;
    const int kb  = kk + hi * 8;
    v16h a  = load_a_cat(hl_row, hr_row, kc0);
    v16h bi = load_b16(Ui + (size_t)col * (2 * H) + kb);
    v16h bo = load_b16(Uo + (size_t)col * (2 * H) + kb);
    v16h bu = load_b16(Uu + (size_t)col * (2 * H) + kb);
    WMMA(acc_i, a, bi);
    WMMA(acc_o, a, bo);
    WMMA(acc_u, a, bu);
  }

  // Forget gates, K = 128 each on hl / hr.
#pragma unroll
  for (int kk = 0; kk < H; kk += 32) {
    const int kc0 = kk + hi * 8;
    const int kb  = kk + hi * 8;
    v16h al = load_a16(hl_row, kc0);
    v16h ar = load_a16(hr_row, kc0);
    v16h b1 = load_b16(F1 + (size_t)col * H + kb);
    v16h b2 = load_b16(F2 + (size_t)col * H + kb);
    WMMA(acc_f1, al, b1);
    WMMA(acc_f2, ar, b2);
  }

#pragma unroll
  for (int r = 0; r < 8; ++r) {
    const int lm = r + 8 * hi;
    const int mm = mbase + lm;
    int mmc = mm < n_out ? mm : (n_out - 1);
    float iv = fsigmoid(acc_i[r]);
    float ov = fsigmoid(acc_o[r]);
    float uv = ftanh(acc_u[r]);
    float f1 = fsigmoid(acc_f1[r]);
    float f2 = fsigmoid(acc_f2[r]);
    float cl = c_in[(size_t)(2 * mmc) * H + col];
    float cr = c_in[(size_t)(2 * mmc + 1) * H + col];
    float cv = iv * uv + f1 * cl + f2 * cr;
    float hv = ov * ftanh(cv);
    if (mm < n_out) {
      c_out[(size_t)mm * H + col] = cv;
      h_out[(size_t)mm * H + col] = (_Float16)hv;
    }
    h_lds[lm * H + col] = hv;
  }
  __syncthreads();

  if (tid < 80) {
    const int node = tid / 5, cls = tid % 5;
    const int mm = mbase + node;
    if (mm < n_out) {
      float s = ob[cls];
      const float* wr = ow + cls * H;
#pragma unroll 8
      for (int k = 0; k < H; ++k) s += h_lds[node * H + k] * wr[k];
      out[(size_t)mm * 5 + cls] = s;
    }
  }
}

// ---- merged tail: levels with n_out = 16, 8, 4, 2, 1 in ONE block ----------
// h/c live in LDS between levels; A-fragments come from LDS (ds_load_b128).
__global__ __launch_bounds__(256) void tree_tail_kernel(
    const _Float16* __restrict__ h_in, const float* __restrict__ c_in,
    float* __restrict__ out,   // already offset to the n_out=16 level's rows
    const _Float16* __restrict__ Ui, const _Float16* __restrict__ Uo,
    const _Float16* __restrict__ Uu,
    const _Float16* __restrict__ F1, const _Float16* __restrict__ F2,
    const float* __restrict__ Ui_b, const float* __restrict__ Uo_b,
    const float* __restrict__ Uu_b,
    const float* __restrict__ F1_b, const float* __restrict__ F2_b,
    const float* __restrict__ ow, const float* __restrict__ ob) {
  __shared__ _Float16 h_buf[32 * H];   // 8 KB
  __shared__ float    c_buf[32 * H];   // 16 KB
  const int tid  = threadIdx.x;
  const int wave = tid >> 5;
  const int lane = tid & 31;
  const int lrow = lane & 15;
  const int hi   = lane >> 4;
  const int col = (wave << 4) + lrow;

  // Stage the 32 input nodes (output of the n_out=32 level) into LDS.
  for (int i = tid; i < 32 * H; i += 256) {
    h_buf[i] = h_in[i];
    c_buf[i] = c_in[i];
  }
  __syncthreads();

  const float bi = Ui_b[col], bo = Uo_b[col], bu = Uu_b[col];
  const float b1 = F1_b[col], b2 = F2_b[col];

  size_t rowoff = 0;
  for (int n_out = 16; n_out >= 1; n_out >>= 1) {
    int mrow = lrow;
    if (mrow >= n_out) mrow = n_out - 1;
    const _Float16* hl_row = &h_buf[(2 * mrow) * H];
    const _Float16* hr_row = hl_row + H;

    v8f acc_i, acc_o, acc_u, acc_f1, acc_f2;
#pragma unroll
    for (int r = 0; r < 8; ++r) {
      acc_i[r] = bi; acc_o[r] = bo; acc_u[r] = bu;
      acc_f1[r] = b1; acc_f2[r] = b2;
    }

#pragma unroll
    for (int kk = 0; kk < 2 * H; kk += 32) {
      const int kc0 = kk + hi * 8;
      const int kb  = kk + hi * 8;
      v16h a  = load_a_cat(hl_row, hr_row, kc0);
      v16h wi = load_b16(Ui + (size_t)col * (2 * H) + kb);
      v16h wo = load_b16(Uo + (size_t)col * (2 * H) + kb);
      v16h wu = load_b16(Uu + (size_t)col * (2 * H) + kb);
      WMMA(acc_i, a, wi);
      WMMA(acc_o, a, wo);
      WMMA(acc_u, a, wu);
    }
#pragma unroll
    for (int kk = 0; kk < H; kk += 32) {
      const int kc0 = kk + hi * 8;
      const int kb  = kk + hi * 8;
      v16h al = load_a16(hl_row, kc0);
      v16h ar = load_a16(hr_row, kc0);
      v16h w1 = load_b16(F1 + (size_t)col * H + kb);
      v16h w2 = load_b16(F2 + (size_t)col * H + kb);
      WMMA(acc_f1, al, w1);
      WMMA(acc_f2, ar, w2);
    }

    // All reads of h_buf/c_buf happen here, before the write barrier.
    float cvs[8], hvs[8];
#pragma unroll
    for (int r = 0; r < 8; ++r) {
      const int mm = r + 8 * hi;
      const int mmc = mm < n_out ? mm : (n_out - 1);
      float iv = fsigmoid(acc_i[r]);
      float ov = fsigmoid(acc_o[r]);
      float uv = ftanh(acc_u[r]);
      float f1 = fsigmoid(acc_f1[r]);
      float f2 = fsigmoid(acc_f2[r]);
      float cl = c_buf[(2 * mmc) * H + col];
      float cr = c_buf[(2 * mmc + 1) * H + col];
      cvs[r] = iv * uv + f1 * cl + f2 * cr;
      hvs[r] = ov * ftanh(cvs[r]);
    }
    __syncthreads();

#pragma unroll
    for (int r = 0; r < 8; ++r) {
      const int mm = r + 8 * hi;
      if (mm < n_out) {
        c_buf[mm * H + col] = cvs[r];
        h_buf[mm * H + col] = (_Float16)hvs[r];
      }
    }
    __syncthreads();

    if (tid < 5 * n_out) {
      const int node = tid / 5, cls = tid % 5;
      float s = ob[cls];
      const float* wr = ow + cls * H;
#pragma unroll 8
      for (int k = 0; k < H; ++k) s += (float)h_buf[node * H + k] * wr[k];
      out[(rowoff + node) * 5 + cls] = s;
    }
    rowoff += (size_t)n_out;
    __syncthreads();   // out-proj reads done before next level's writes
  }
}

// ---- host -------------------------------------------------------------------

extern "C" void kernel_launch(void* const* d_in, const int* in_sizes, int n_in,
                              void* d_out, int out_size, void* d_ws, size_t ws_size,
                              hipStream_t stream) {
  const int*   tokens = (const int*)d_in[0];
  const int*   labels = (const int*)d_in[1];
  const float* embed  = (const float*)d_in[2];
  const float* W_i_w = (const float*)d_in[3];  const float* W_i_b = (const float*)d_in[4];
  const float* W_o_w = (const float*)d_in[5];  const float* W_o_b = (const float*)d_in[6];
  const float* W_u_w = (const float*)d_in[7];  const float* W_u_b = (const float*)d_in[8];
  const float* U_i_w = (const float*)d_in[9];  const float* U_i_b = (const float*)d_in[10];
  const float* U_o_w = (const float*)d_in[11]; const float* U_o_b = (const float*)d_in[12];
  const float* U_u_w = (const float*)d_in[13]; const float* U_u_b = (const float*)d_in[14];
  const float* F1_w  = (const float*)d_in[15]; const float* F1_b  = (const float*)d_in[16];
  const float* F2_w  = (const float*)d_in[17]; const float* F2_b  = (const float*)d_in[18];
  const float* out_w = (const float*)d_in[19]; const float* out_b = (const float*)d_in[20];
  float* out = (float*)d_out;

  char* ws = (char*)d_ws;
  size_t off = 0;
  auto carve = [&](size_t bytes) {
    size_t o = off;
    off = (off + bytes + 255) & ~(size_t)255;
    return o;
  };
  _Float16* h0 = (_Float16*)(ws + carve((size_t)NL * H * 2));
  _Float16* h1 = (_Float16*)(ws + carve((size_t)NL * H * 2));
  float*    c0 = (float*)   (ws + carve((size_t)NL * H * 4));
  float*    c1 = (float*)   (ws + carve((size_t)NL * H * 4));
  _Float16* Wi16 = (_Float16*)(ws + carve((size_t)H * H * 2));
  _Float16* Wo16 = (_Float16*)(ws + carve((size_t)H * H * 2));
  _Float16* Wu16 = (_Float16*)(ws + carve((size_t)H * H * 2));
  _Float16* F116 = (_Float16*)(ws + carve((size_t)H * H * 2));
  _Float16* F216 = (_Float16*)(ws + carve((size_t)H * H * 2));
  _Float16* Ui16 = (_Float16*)(ws + carve((size_t)H * 2 * H * 2));
  _Float16* Uo16 = (_Float16*)(ws + carve((size_t)H * 2 * H * 2));
  _Float16* Uu16 = (_Float16*)(ws + carve((size_t)H * 2 * H * 2));

  // Convert weights to f16 (small, once per call).
  const int nHH = H * H, nH2H = H * 2 * H;
  cvt_f32_f16_kernel<<<(nHH + 255) / 256, 256, 0, stream>>>(W_i_w, Wi16, nHH);
  cvt_f32_f16_kernel<<<(nHH + 255) / 256, 256, 0, stream>>>(W_o_w, Wo16, nHH);
  cvt_f32_f16_kernel<<<(nHH + 255) / 256, 256, 0, stream>>>(W_u_w, Wu16, nHH);
  cvt_f32_f16_kernel<<<(nHH + 255) / 256, 256, 0, stream>>>(F1_w,  F116, nHH);
  cvt_f32_f16_kernel<<<(nHH + 255) / 256, 256, 0, stream>>>(F2_w,  F216, nHH);
  cvt_f32_f16_kernel<<<(nH2H + 255) / 256, 256, 0, stream>>>(U_i_w, Ui16, nH2H);
  cvt_f32_f16_kernel<<<(nH2H + 255) / 256, 256, 0, stream>>>(U_o_w, Uo16, nH2H);
  cvt_f32_f16_kernel<<<(nH2H + 255) / 256, 256, 0, stream>>>(U_u_w, Uu16, nH2H);

  // Leaf level: rows [0, 65536) of the output.
  tree_leaf_kernel<<<NL / 16, 256, 0, stream>>>(
      tokens, embed, h0, c0, out,
      Wi16, Wo16, Wu16, W_i_b, W_o_b, W_u_b, out_w, out_b, NL);

  // Big merge levels: n_out = 32768 ... 32  (l = 1..11).
  _Float16* hb[2] = {h0, h1};
  float*    cb[2] = {c0, c1};
  int cur = 0;
  int n = NL;
  size_t rowoff = NL;
  for (int l = 1; l <= 11; ++l) {
    int n_out = n >> 1;
    int grid = (n_out + 15) / 16;
    tree_level_kernel<<<grid, 256, 0, stream>>>(
        hb[cur], cb[cur], hb[1 - cur], cb[1 - cur],
        out + rowoff * 5,
        Ui16, Uo16, Uu16, F116, F216,
        U_i_b, U_o_b, U_u_b, F1_b, F2_b,
        out_w, out_b, n_out);
    rowoff += (size_t)n_out;
    n = n_out;
    cur ^= 1;
  }

  // Merged tail: n_out = 16, 8, 4, 2, 1 in a single block (rows 131040..131070).
  tree_tail_kernel<<<1, 256, 0, stream>>>(
      hb[cur], cb[cur], out + rowoff * 5,
      Ui16, Uo16, Uu16, F116, F216,
      U_i_b, U_o_b, U_u_b, F1_b, F2_b, out_w, out_b);

  // Second tuple element: node_labels appended after the 131071x5 outputs.
  const int n_labels = in_sizes[1];
  copy_labels_kernel<<<(n_labels + 255) / 256, 256, 0, stream>>>(
      labels, out + (size_t)131071 * 5, n_labels);
}
